// NeighborEncoder_13606456394604
// MI455X (gfx1250) — matrix operand order repaired and checked by reading, not compiled, not run on hardware
//
#include <hip/hip_runtime.h>

typedef __attribute__((ext_vector_type(16))) _Float16 v16h;
typedef __attribute__((ext_vector_type(8)))  _Float16 v8h;
typedef __attribute__((ext_vector_type(4)))  _Float16 v4h;
typedef __attribute__((ext_vector_type(8)))  float    v8f;

#define NBK   32     // NB blocks per sample (softmax axis)
#define HD    128    // H
#define NVK   4      // NV
#define ATTN  64     // ATTN_DIM
#define KDIM  256    // 2*H
#define AS    264    // padded LDS row stride (f16 elems), 528B keeps rows 16B-aligned

__device__ __forceinline__ float fast_tanh(float x) {
#if __has_builtin(__builtin_amdgcn_tanhf)
    return __builtin_amdgcn_tanhf(x);      // v_tanh_f32 (gfx1250 trans op)
#else
    return tanhf(x);
#endif
}

// --- tiny one-off: fc (256x64 f32, row-major) -> fcT (64x256 f16, transposed) ---
__global__ void cvt_fc_kernel(const float* __restrict__ fc, _Float16* __restrict__ fcT) {
    int idx = blockIdx.x * blockDim.x + threadIdx.x;   // 0 .. 64*256-1
    int col = idx >> 8;                                // 0..63  (ATTN col)
    int k   = idx & 255;                               // 0..255 (K)
    fcT[idx] = (_Float16)fc[k * ATTN + col];           // fcT[col*256 + k]
}

__global__ __launch_bounds__(256)
void neigh_attn_kernel(const float* __restrict__ inputs,   // (N, 32, 128)
                       const float* __restrict__ neigh,    // (N*4, 32, 128) flat
                       const _Float16* __restrict__ fcT,   // (64, 256) f16
                       const float* __restrict__ tw,       // (64)
                       float* __restrict__ out)            // (N*4, 128)
{
    __shared__ _Float16 sA[NBK * AS];   // cat matrix 32 x 256 in f16 (padded)
    __shared__ float sScore[NBK];
    __shared__ float sAttn[NBK];

    const int s   = blockIdx.x;          // sample in [0, N*NV)
    const int n   = s >> 2;              // NV == 4 (jnp.repeat semantics)
    const int tid = threadIdx.x;

    if (tid < NBK) sScore[tid] = 0.f;

    // ---- stage cat = [inputs[n] | neigh[s]] as f16 into LDS (coalesced float4) ----
    const float* inp_base = inputs + (size_t)n * NBK * HD;
    const float* ngh_base = neigh  + (size_t)s * NBK * HD;
    #pragma unroll
    for (int it = 0; it < 8; ++it) {
        int flat = (tid + it * 256) * 4;     // element index in 32x256
        int row  = flat >> 8;
        int col  = flat & 255;
        float4 v = (col < HD) ? *(const float4*)(inp_base + row * HD + col)
                              : *(const float4*)(ngh_base + row * HD + (col - HD));
        v4h h; h.x = (_Float16)v.x; h.y = (_Float16)v.y;
               h.z = (_Float16)v.z; h.w = (_Float16)v.w;
        *(v4h*)(&sA[row * AS + col]) = h;
    }
    __syncthreads();

    // ---- proj(32x64) = cat(32x256) @ fc(256x64) via 8 x v_wmma_f32_16x16x32_f16 ----
    const int wave = tid >> 5;           // 8 waves -> 8 output tiles
    const int lane = tid & 31;
    const int lm   = lane & 15;
    const int lg   = lane >> 4;          // half-wave group
    const int mt   = wave & 1;           // M tile: rows mt*16..+15
    const int nt   = wave >> 1;          // N tile: cols nt*16..+15

    v8f c = {};
    const int arow = mt * 16 + lm;
    const _Float16* bcol = fcT + (nt * 16 + lm) * KDIM;   // one column's K-run
    #pragma unroll
    for (int kk = 0; kk < 8; ++kk) {
        int kb = kk * 32 + lg * 8;
        v8h alo = *(const v8h*)(&sA[arow * AS + kb]);        // K = kb..kb+7
        v8h ahi = *(const v8h*)(&sA[arow * AS + kb + 16]);   // K = kb+16..kb+23
        v16h a  = __builtin_shufflevector(alo, ahi,
                     0,1,2,3,4,5,6,7,8,9,10,11,12,13,14,15);
        v8h blo = *(const v8h*)(bcol + kb);                  // K = kb..kb+7
        v8h bhi = *(const v8h*)(bcol + kb + 16);             // K = kb+16..kb+23
        v16h b  = __builtin_shufflevector(blo, bhi,
                     0,1,2,3,4,5,6,7,8,9,10,11,12,13,14,15);
        c = __builtin_amdgcn_wmma_f32_16x16x32_f16(
                false, a, false, b, (short)0, c, false, false);
    }

    // ---- scores[row] = sum_col tanh(proj[row,col]) * tw[col] ----
    // lane lm holds column nt*16+lm, rows lg*8+r  -> 16-lane shuffle reduce
    float twl = tw[nt * 16 + lm];
    #pragma unroll
    for (int r = 0; r < 8; ++r) {
        float v = fast_tanh(c[r]) * twl;
        v += __shfl_xor(v, 1, 32);
        v += __shfl_xor(v, 2, 32);
        v += __shfl_xor(v, 4, 32);
        v += __shfl_xor(v, 8, 32);
        if (lm == 0) atomicAdd(&sScore[mt * 16 + lg * 8 + r], v);
    }
    __syncthreads();

    // ---- softmax over the 32 rows (wave 0) ----
    if (tid < NBK) {
        float x = sScore[tid];
        float m = x;
        #pragma unroll
        for (int off = 16; off > 0; off >>= 1) m = fmaxf(m, __shfl_xor(m, off, 32));
        float e = __expf(x - m);
        float sm = e;
        #pragma unroll
        for (int off = 16; off > 0; off >>= 1) sm += __shfl_xor(sm, off, 32);
        sAttn[tid] = e / sm;
    }
    __syncthreads();

    // ---- agg[h] = sum_b attn[b]*neigh[b,h] (f16 neigh already in LDS), then ELU ----
    if (tid < HD) {
        float acc = 0.f;
        #pragma unroll
        for (int b = 0; b < NBK; ++b)
            acc += sAttn[b] * (float)sA[b * AS + HD + tid];
        out[(size_t)s * HD + tid] = (acc > 0.f) ? acc : (__expf(acc) - 1.f);
    }
}

extern "C" void kernel_launch(void* const* d_in, const int* in_sizes, int n_in,
                              void* d_out, int out_size, void* d_ws, size_t ws_size,
                              hipStream_t stream) {
    const float* inputs = (const float*)d_in[0];   // (N, 32, 128)
    const float* neigh  = (const float*)d_in[1];   // (N, 4, 32, 128)
    const float* fc     = (const float*)d_in[2];   // (256, 64)
    const float* tw     = (const float*)d_in[3];   // (64, 1)
    float* out = (float*)d_out;

    _Float16* fcT = (_Float16*)d_ws;               // 64*256 f16 = 32 KB scratch

    cvt_fc_kernel<<<(ATTN * KDIM) / 256, 256, 0, stream>>>(fc, fcT);

    int S = in_sizes[1] / (NBK * HD);              // N*NV = 40000
    neigh_attn_kernel<<<S, 256, 0, stream>>>(inputs, neigh, fcT, tw, out);
}
